// Actor_57526791963315
// MI455X (gfx1250) — matrix operand order repaired and checked by reading, not compiled, not run on hardware
//
#include <hip/hip_runtime.h>
#include <hip/hip_bf16.h>
#include <cstdint>

// ---------------------------------------------------------------------------
// Problem constants
// ---------------------------------------------------------------------------
#define B_    2048
#define VMAX_ 128
#define EGO_  15
#define H_    256
#define ACT_  2

typedef __attribute__((ext_vector_type(16))) _Float16 v16h;
typedef __attribute__((ext_vector_type(8)))  float    v8f;

union F16x16 { v16h v; uint4 q[2]; };
union F32x8  { v8f  v; float f[8]; };

__device__ __forceinline__ float sigmf(float x) { return 1.0f / (1.0f + __expf(-x)); }

// Produce a zero whose value the compiler cannot prove constant across loop
// iterations. Adding it to load indices pins those loads inside the loop
// (prevents LICM from keeping 32-36 loop-invariant B-fragments live -> spills)
// while keeping the base pointer a pristine kernel arg, so the loads lower
// to global_load (saddr + 32-bit voffset) instead of flat_load.
__device__ __forceinline__ int launder_zero() {
  int z = 0;
  asm volatile("" : "+s"(z));
  return z;
}

// A fragment (16x32 f16, row-major source, element stride `stride`).
// lane: M = lane&15, half = lane>>4 ; VGPR v holds K pair
//   k = (v/4)*16 + half*8 + (v%4)*2  -> two contiguous 16B chunks per lane.
__device__ __forceinline__ v16h load_a_f16(const _Float16* tile, int stride, int lane) {
  const _Float16* p = tile + (lane & 15) * stride + (((lane >> 4) & 1) << 3);
  F16x16 r;
  r.q[0] = *(const uint4*)(p);        // k = kh*8 .. kh*8+7
  r.q[1] = *(const uint4*)(p + 16);   // k = 16 + kh*8 ..
  return r.v;
}

// B fragment (32x16): A @ W^T with W row-major [N][K] f16, B(k,n)=W[n][k].
// lane: N = n0 + (lane&15); K = k0 + (lane>>4)*8 + {0..7, 16..23}.
__device__ __forceinline__ v16h load_b_f16(const _Float16* W, int n0, int k0, int K, int lane) {
  const _Float16* p = W + (size_t)(n0 + (lane & 15)) * K + k0 + (((lane >> 4) & 1) << 3);
  F16x16 r;
  r.q[0] = *(const uint4*)(p);
  r.q[1] = *(const uint4*)(p + 16);
  return r.v;
}

__device__ __forceinline__ v8f wmma16(v16h a, v16h b, v8f c) {
  return __builtin_amdgcn_wmma_f32_16x16x32_f16(false, a, false, b, (short)0, c, false, false);
}

// ---------------------------------------------------------------------------
// Prep: f32 -> f16 weight conversion (+pad Wih1 K:15->32), fused biases
// ---------------------------------------------------------------------------
__global__ __launch_bounds__(256) void prep_kernel(
    const float* __restrict__ Wih1, const float* __restrict__ Whh1,
    const float* __restrict__ bih1, const float* __restrict__ bhh1,
    const float* __restrict__ W1, const float* __restrict__ W2,
    const float* __restrict__ W3, const float* __restrict__ W4,
    const float* __restrict__ Wih2, const float* __restrict__ Whh2,
    const float* __restrict__ bih2, const float* __restrict__ bhh2,
    _Float16* __restrict__ Wih1h, _Float16* __restrict__ Whh1h,
    _Float16* __restrict__ W1h, _Float16* __restrict__ W2h,
    _Float16* __restrict__ W3h, _Float16* __restrict__ W4h,
    _Float16* __restrict__ Wih2h, _Float16* __restrict__ Whh2h,
    float* __restrict__ benc, float* __restrict__ bdec)
{
  const int stride = gridDim.x * blockDim.x;
  const int t0 = blockIdx.x * blockDim.x + threadIdx.x;
  for (int i = t0; i < 1024 * 32; i += stride) {
    int n = i >> 5, k = i & 31;
    Wih1h[i] = (_Float16)((k < EGO_) ? Wih1[n * EGO_ + k] : 0.0f);
  }
  for (int i = t0; i < 1024 * 256; i += stride) Whh1h[i] = (_Float16)Whh1[i];
  for (int i = t0; i < 1024 * 256; i += stride) W1h[i]   = (_Float16)W1[i];
  for (int i = t0; i < 1024 * 1024; i += stride) W2h[i]  = (_Float16)W2[i];
  for (int i = t0; i < 512 * 1024; i += stride) W3h[i]   = (_Float16)W3[i];
  for (int i = t0; i < 256 * 512; i += stride) W4h[i]    = (_Float16)W4[i];
  for (int i = t0; i < 1024 * 256; i += stride) Wih2h[i] = (_Float16)Wih2[i];
  for (int i = t0; i < 1024 * 256; i += stride) Whh2h[i] = (_Float16)Whh2[i];
  for (int i = t0; i < 1024; i += stride) {
    benc[i] = bih1[i] + bhh1[i];
    bdec[i] = bih2[i] + bhh2[i];
  }
}

// ---------------------------------------------------------------------------
// Encoder LSTM: 64 WGs x 32 batch rows, 16 waves, 128 steps, no global sync.
// Each wave owns 16 h-columns across all 4 gates -> 8 C-tiles (64 VGPRs).
// ---------------------------------------------------------------------------
__global__ __launch_bounds__(512) void enc_kernel(
    const float* __restrict__ state, const int* __restrict__ lengths,
    const _Float16* __restrict__ Whh, const _Float16* __restrict__ Wih,
    const float* __restrict__ benc, float* __restrict__ henc)
{
  __shared__ __align__(16) _Float16 hbuf[32 * 256];   // h state, f16
  __shared__ __align__(16) _Float16 xbuf[32 * 32];    // x_t padded to K=32
  __shared__ float bbuf[1024];
  __shared__ int   lenbuf[32];

  const int tid  = threadIdx.x;
  const int lane = tid & 31;
  const int w    = tid >> 5;          // 0..15
  const int rb   = blockIdx.x * 32;

  for (int i = tid; i < 32 * 256; i += 512) hbuf[i] = (_Float16)0.0f;
  for (int i = tid; i < 1024; i += 512) bbuf[i] = benc[i];
  if (tid < 32) lenbuf[tid] = lengths[rb + tid];
  for (int i = tid; i < 32 * 32; i += 512) {
    int m = i >> 5, e = i & 31;
    xbuf[i] = (_Float16)((e < EGO_) ? state[((size_t)(rb + m) * VMAX_ + 0) * EGO_ + e] : 0.0f);
  }
  __syncthreads();

  const int kh   = (lane >> 4) & 1;
  const int ln   = lane & 15;
  const int hcol = w * 16 + ln;

  // loop-invariant per-lane scalars
  const float bb0 = bbuf[hcol];
  const float bb1 = bbuf[256 + hcol];
  const float bb2 = bbuf[512 + hcol];
  const float bb3 = bbuf[768 + hcol];
  int lenr[16];
#pragma unroll
  for (int rt = 0; rt < 2; ++rt)
#pragma unroll
    for (int r = 0; r < 8; ++r) lenr[rt * 8 + r] = lenbuf[rt * 16 + kh * 8 + r];

  float creg[2][8], hreg[2][8];
#pragma unroll
  for (int rt = 0; rt < 2; ++rt)
#pragma unroll
    for (int r = 0; r < 8; ++r) { creg[rt][r] = 0.0f; hreg[rt][r] = 0.0f; }

#pragma unroll 1
  for (int t = 0; t < VMAX_; ++t) {
    // per-iteration opaque zero: pins weight loads inside the loop while
    // keeping the base pointer global (saddr-form global_load_b128)
    const int z = launder_zero();

    F32x8 acc[8];                       // [gate*2 + rowtile]
#pragma unroll
    for (int i = 0; i < 8; ++i)
#pragma unroll
      for (int r = 0; r < 8; ++r) acc[i].f[r] = 0.0f;

    // gates += h @ Whh^T
#pragma unroll
    for (int ks = 0; ks < 8; ++ks) {
      v16h a0 = load_a_f16(hbuf + ks * 32, 256, lane);
      v16h a1 = load_a_f16(hbuf + 16 * 256 + ks * 32, 256, lane);
#pragma unroll
      for (int g = 0; g < 4; ++g) {
        v16h b = load_b_f16(Whh, g * 256 + w * 16, ks * 32 + z, 256, lane);
        acc[g * 2].v     = wmma16(a0, b, acc[g * 2].v);
        acc[g * 2 + 1].v = wmma16(a1, b, acc[g * 2 + 1].v);
      }
    }
    // gates += x_t @ Wih^T (K padded to 32)
    {
      v16h a0 = load_a_f16(xbuf, 32, lane);
      v16h a1 = load_a_f16(xbuf + 16 * 32, 32, lane);
#pragma unroll
      for (int g = 0; g < 4; ++g) {
        v16h b = load_b_f16(Wih, g * 256 + w * 16, z, 32, lane);
        acc[g * 2].v     = wmma16(a0, b, acc[g * 2].v);
        acc[g * 2 + 1].v = wmma16(a1, b, acc[g * 2 + 1].v);
      }
    }

    // masked LSTM pointwise update (gate order: i, f, g, o)
#pragma unroll
    for (int rt = 0; rt < 2; ++rt)
#pragma unroll
      for (int r = 0; r < 8; ++r) {
        float iv = acc[0 + rt].f[r] + bb0;
        float fv = acc[2 + rt].f[r] + bb1;
        float gv = acc[4 + rt].f[r] + bb2;
        float ov = acc[6 + rt].f[r] + bb3;
        float co = creg[rt][r];
        float cn = sigmf(fv) * co + sigmf(iv) * tanhf(gv);
        float hn = sigmf(ov) * tanhf(cn);
        bool  mk = (t < lenr[rt * 8 + r]);
        creg[rt][r] = mk ? cn : co;
        hreg[rt][r] = mk ? hn : hreg[rt][r];
      }
    __syncthreads();   // all reads of hbuf/xbuf complete
#pragma unroll
    for (int rt = 0; rt < 2; ++rt)
#pragma unroll
      for (int r = 0; r < 8; ++r)
        hbuf[(rt * 16 + kh * 8 + r) * 256 + hcol] = (_Float16)hreg[rt][r];
    if (t + 1 < VMAX_) {
      for (int i = tid; i < 32 * 32; i += 512) {
        int m = i >> 5, e = i & 31;
        xbuf[i] = (_Float16)((e < EGO_)
                    ? state[((size_t)(rb + m) * VMAX_ + (t + 1)) * EGO_ + e] : 0.0f);
      }
    }
    __syncthreads();   // writes visible before next step's GEMM
  }

  for (int i = tid; i < 32 * 256; i += 512)
    henc[(size_t)(rb + (i >> 8)) * 256 + (i & 255)] = (float)hbuf[i];
}

// ---------------------------------------------------------------------------
// Generic WMMA GEMM: out = act(A @ W^T + bias), A f32 [M,K], W f16 [N,K].
// Block tile 32 rows x 128 cols; 8 waves x (2 row-tiles x 1 col-tile).
// ---------------------------------------------------------------------------
template <bool RELU>
__global__ __launch_bounds__(256) void gemm_kernel(
    const float* __restrict__ A, const _Float16* __restrict__ W,
    const float* __restrict__ bias, float* __restrict__ out,
    int M, int N, int K)
{
  __shared__ __align__(16) _Float16 As[32 * 32];
  const int tid  = threadIdx.x;
  const int lane = tid & 31;
  const int w    = tid >> 5;
  const int rowb = blockIdx.y * 32;
  const int colb = blockIdx.x * 128 + w * 16;

  F32x8 acc[2];
#pragma unroll
  for (int i = 0; i < 2; ++i)
#pragma unroll
    for (int r = 0; r < 8; ++r) acc[i].f[r] = 0.0f;

  for (int kb = 0; kb < K; kb += 32) {
    for (int i = tid; i < 32 * 32; i += 256) {
      int m = i >> 5, k = i & 31;
      As[i] = (_Float16)A[(size_t)(rowb + m) * K + kb + k];
    }
    __syncthreads();
    v16h a0 = load_a_f16(As, 32, lane);
    v16h a1 = load_a_f16(As + 16 * 32, 32, lane);
    v16h b  = load_b_f16(W, colb, kb, K, lane);
    acc[0].v = wmma16(a0, b, acc[0].v);
    acc[1].v = wmma16(a1, b, acc[1].v);
    __syncthreads();
  }

  const int n = colb + (lane & 15);
  const float bv = bias[n];
  const int kh = (lane >> 4) & 1;
#pragma unroll
  for (int rt = 0; rt < 2; ++rt)
#pragma unroll
    for (int r = 0; r < 8; ++r) {
      int m = rowb + rt * 16 + kh * 8 + r;
      float v = acc[rt].f[r] + bv;
      if (RELU) v = fmaxf(v, 0.0f);
      out[(size_t)m * N + n] = v;
    }
}

// ---------------------------------------------------------------------------
// Decoder LSTM: gates = xw (precomputed, constant over t) + h @ Whh2^T.
// Action head: act = tanh(c_new @ Wpi^T + bpi), masked by lengths.
// ---------------------------------------------------------------------------
__global__ __launch_bounds__(512) void dec_kernel(
    const float* __restrict__ xw, const _Float16* __restrict__ Whh,
    const float* __restrict__ Wpi, const float* __restrict__ bpi,
    const int* __restrict__ lengths, float* __restrict__ out)
{
  __shared__ __align__(16) _Float16 hbuf[32 * 256];
  __shared__ float cbuf[32 * 256];
  __shared__ float wpib[ACT_ * 256];
  __shared__ float bpib[ACT_];
  __shared__ int   lenbuf[32];

  const int tid  = threadIdx.x;
  const int lane = tid & 31;
  const int w    = tid >> 5;          // 0..15
  const int rb   = blockIdx.x * 32;

  for (int i = tid; i < 32 * 256; i += 512) hbuf[i] = (_Float16)0.0f;
  for (int i = tid; i < ACT_ * 256; i += 512) wpib[i] = Wpi[i];
  if (tid < ACT_) bpib[tid] = bpi[tid];
  if (tid < 32) lenbuf[tid] = lengths[rb + tid];
  __syncthreads();

  const int kh   = (lane >> 4) & 1;
  const int ln   = lane & 15;
  const int hcol = w * 16 + ln;
  const float* xwp = xw + (size_t)rb * 1024 + hcol;

  float creg[2][8];
#pragma unroll
  for (int rt = 0; rt < 2; ++rt)
#pragma unroll
    for (int r = 0; r < 8; ++r) creg[rt][r] = 0.0f;

  // action-head thread mapping: 64 outputs, 8 lanes each
  const int pair = tid >> 3;   // 0..63
  const int sub  = tid & 7;
  const int am   = pair >> 1;  // batch row 0..31
  const int aa   = pair & 1;   // action idx
  const int alen = lenbuf[am];
  const float ab = bpib[aa];

#pragma unroll 1
  for (int t = 0; t < VMAX_; ++t) {
    const int z = launder_zero();

    F32x8 acc[8];
    // init gates from precomputed xw (includes bih2+bhh2)
#pragma unroll
    for (int g = 0; g < 4; ++g)
#pragma unroll
      for (int rt = 0; rt < 2; ++rt)
#pragma unroll
        for (int r = 0; r < 8; ++r) {
          const int ml = rt * 16 + kh * 8 + r;
          acc[g * 2 + rt].f[r] = xwp[(size_t)(ml * 1024 + g * 256 + z)];
        }
    // gates += h @ Whh2^T
#pragma unroll
    for (int ks = 0; ks < 8; ++ks) {
      v16h a0 = load_a_f16(hbuf + ks * 32, 256, lane);
      v16h a1 = load_a_f16(hbuf + 16 * 256 + ks * 32, 256, lane);
#pragma unroll
      for (int g = 0; g < 4; ++g) {
        v16h b = load_b_f16(Whh, g * 256 + w * 16, ks * 32 + z, 256, lane);
        acc[g * 2].v     = wmma16(a0, b, acc[g * 2].v);
        acc[g * 2 + 1].v = wmma16(a1, b, acc[g * 2 + 1].v);
      }
    }

    float hstore[2][8];
#pragma unroll
    for (int rt = 0; rt < 2; ++rt)
#pragma unroll
      for (int r = 0; r < 8; ++r) {
        float iv = acc[0 + rt].f[r];
        float fv = acc[2 + rt].f[r];
        float gv = acc[4 + rt].f[r];
        float ov = acc[6 + rt].f[r];
        float cn = sigmf(fv) * creg[rt][r] + sigmf(iv) * tanhf(gv);
        creg[rt][r]   = cn;
        hstore[rt][r] = sigmf(ov) * tanhf(cn);
      }
    __syncthreads();   // reads of hbuf / cbuf done
#pragma unroll
    for (int rt = 0; rt < 2; ++rt)
#pragma unroll
      for (int r = 0; r < 8; ++r) {
        const int ml = rt * 16 + kh * 8 + r;
        hbuf[ml * 256 + hcol] = (_Float16)hstore[rt][r];
        cbuf[ml * 256 + hcol] = creg[rt][r];
      }
    __syncthreads();   // h/c visible

    // action head: act[m][a] = tanh(dot(c[m,:], Wpi[a,:]) + bpi[a])
    {
      float s = 0.0f;
      const float* crow = cbuf + am * 256 + sub * 32;
      const float* wrow = wpib + aa * 256 + sub * 32;
#pragma unroll 8
      for (int k = 0; k < 32; ++k) s += crow[k] * wrow[k];
      s += __shfl_xor(s, 1);
      s += __shfl_xor(s, 2);
      s += __shfl_xor(s, 4);
      if (sub == 0) {
        bool mk = (t < alen);
        out[((size_t)(rb + am) * VMAX_ + t) * ACT_ + aa] = mk ? tanhf(s + ab) : 0.0f;
      }
    }
  }
}

// ---------------------------------------------------------------------------
// Host launch
// ---------------------------------------------------------------------------
extern "C" void kernel_launch(void* const* d_in, const int* in_sizes, int n_in,
                              void* d_out, int out_size, void* d_ws, size_t ws_size,
                              hipStream_t stream) {
  (void)in_sizes; (void)n_in; (void)out_size; (void)ws_size;

  const float* state = (const float*)d_in[0];
  const int*   lengths = (const int*)d_in[1];
  const float* Wih1 = (const float*)d_in[2];
  const float* Whh1 = (const float*)d_in[3];
  const float* bih1 = (const float*)d_in[4];
  const float* bhh1 = (const float*)d_in[5];
  const float* W1 = (const float*)d_in[6];
  const float* b1 = (const float*)d_in[7];
  const float* W2 = (const float*)d_in[8];
  const float* b2 = (const float*)d_in[9];
  const float* W3 = (const float*)d_in[10];
  const float* b3 = (const float*)d_in[11];
  const float* W4 = (const float*)d_in[12];
  const float* b4 = (const float*)d_in[13];
  const float* Wih2 = (const float*)d_in[14];
  const float* Whh2 = (const float*)d_in[15];
  const float* bih2 = (const float*)d_in[16];
  const float* bhh2 = (const float*)d_in[17];
  const float* Wpi = (const float*)d_in[18];
  const float* bpi = (const float*)d_in[19];
  float* out = (float*)d_out;

  char* ws = (char*)d_ws;
  size_t off = 0;
  auto alloc = [&](size_t bytes) -> void* {
    void* p = ws + off; off += (bytes + 255) & ~(size_t)255; return p;
  };
  _Float16* Wih1h = (_Float16*)alloc((size_t)1024 * 32 * 2);
  _Float16* Whh1h = (_Float16*)alloc((size_t)1024 * 256 * 2);
  _Float16* W1h   = (_Float16*)alloc((size_t)1024 * 256 * 2);
  _Float16* W2h   = (_Float16*)alloc((size_t)1024 * 1024 * 2);
  _Float16* W3h   = (_Float16*)alloc((size_t)512 * 1024 * 2);
  _Float16* W4h   = (_Float16*)alloc((size_t)256 * 512 * 2);
  _Float16* Wih2h = (_Float16*)alloc((size_t)1024 * 256 * 2);
  _Float16* Whh2h = (_Float16*)alloc((size_t)1024 * 256 * 2);
  float* benc = (float*)alloc(1024 * 4);
  float* bdec = (float*)alloc(1024 * 4);
  float* henc = (float*)alloc((size_t)B_ * 256 * 4);
  float* bufA = (float*)alloc((size_t)B_ * 1024 * 4);
  float* bufB = (float*)alloc((size_t)B_ * 1024 * 4);
  float* xw   = (float*)alloc((size_t)B_ * 1024 * 4);

  prep_kernel<<<512, 256, 0, stream>>>(Wih1, Whh1, bih1, bhh1, W1, W2, W3, W4,
                                       Wih2, Whh2, bih2, bhh2,
                                       Wih1h, Whh1h, W1h, W2h, W3h, W4h,
                                       Wih2h, Whh2h, benc, bdec);

  enc_kernel<<<B_ / 32, 512, 0, stream>>>(state, lengths, Whh1h, Wih1h, benc, henc);

  gemm_kernel<true ><<<dim3(1024 / 128, B_ / 32), 256, 0, stream>>>(henc, W1h, b1, bufA, B_, 1024, 256);
  gemm_kernel<true ><<<dim3(1024 / 128, B_ / 32), 256, 0, stream>>>(bufA, W2h, b2, bufB, B_, 1024, 1024);
  gemm_kernel<true ><<<dim3( 512 / 128, B_ / 32), 256, 0, stream>>>(bufB, W3h, b3, bufA, B_, 512, 1024);
  gemm_kernel<true ><<<dim3( 256 / 128, B_ / 32), 256, 0, stream>>>(bufA, W4h, b4, bufB, B_, 256, 512);
  // xw = x4 @ Wih2^T + (bih2 + bhh2) — decoder input term is constant over t
  gemm_kernel<false><<<dim3(1024 / 128, B_ / 32), 256, 0, stream>>>(bufB, Wih2h, bdec, xw, B_, 1024, 256);

  dec_kernel<<<B_ / 32, 512, 0, stream>>>(xw, Whh2h, Wpi, bpi, lengths, out);
}
